// KANDecoder_44066364457154
// MI455X (gfx1250) — matrix operand order repaired and checked by reading, not compiled, not run on hardware
//
#include <hip/hip_runtime.h>
#include <hip/hip_bf16.h>

// ---------------------------------------------------------------------------
// Types for CDNA5 WMMA (wave32, 16x16x32 bf16 -> f32)
// ---------------------------------------------------------------------------
typedef __attribute__((ext_vector_type(16))) __bf16 bf16x16;
typedef __attribute__((ext_vector_type(8)))  float  f32x8;

union BfPack { uint4 u[2]; bf16x16 v; };

__device__ __forceinline__ unsigned short f32_to_bf16_rne(float f) {
    unsigned u = __builtin_bit_cast(unsigned, f);
    u += 0x7FFFu + ((u >> 16) & 1u);          // round-to-nearest-even
    return (unsigned short)(u >> 16);
}
__device__ __forceinline__ float bf16_to_f32(unsigned short h) {
    unsigned u = ((unsigned)h) << 16;
    return __builtin_bit_cast(float, u);
}

// ---------------------------------------------------------------------------
// Split fp32 -> (hi, lo) bf16 pair:  x ~= hi + lo  (~16 mantissa bits)
// ---------------------------------------------------------------------------
__global__ void kd_split_kernel(const float* __restrict__ src,
                                unsigned short* __restrict__ hi,
                                unsigned short* __restrict__ lo, long n) {
    long i = (long)blockIdx.x * blockDim.x + threadIdx.x;
    if (i >= n) return;
    float x = src[i];
    unsigned short h = f32_to_bf16_rne(x);
    hi[i] = h;
    lo[i] = f32_to_bf16_rne(x - bf16_to_f32(h));
}

// ---------------------------------------------------------------------------
// GEMM: C[M,N] = A[M,K] @ W[N,K]^T (+bias) (+residual), bf16x3 split inputs,
// fp32 accumulation on v_wmma_f32_16x16x32_bf16.
// 2x2 register tiling: one 32x32 output block per wave (4 accumulators),
// each A/B fragment reused across two WMMA tiles -> 12 WMMAs / 16 b128 loads
// per 32-deep k-step. All dims are multiples of 32 (no edge handling).
// ---------------------------------------------------------------------------
__global__ void kd_gemm_bf16x3(const unsigned short* __restrict__ Ahi,
                               const unsigned short* __restrict__ Alo,
                               const unsigned short* __restrict__ Whi,
                               const unsigned short* __restrict__ Wlo,
                               const float* __restrict__ bias,
                               const float* __restrict__ residual,
                               float* __restrict__ C,
                               int M, int N, int K) {
    const int wave   = (int)((blockIdx.x * blockDim.x + threadIdx.x) >> 5);
    const int lane   = threadIdx.x & 31;
    const int blocksN = N >> 5;               // 32-wide output blocks
    const int blocksM = M >> 5;
    const int bm = wave / blocksN;
    const int bn = wave - bm * blocksN;
    if (bm >= blocksM) return;                // uniform across the wave

    const int half = lane >> 4;               // 0: lanes 0-15, 1: lanes 16-31
    const int l15  = lane & 15;

    const long arow0 = (long)(bm * 32 + l15) * K;        // A rows (two M-tiles)
    const long arow1 = arow0 + (long)16 * K;
    const long wrow0 = (long)(bn * 32 + l15) * K;        // W rows (two N-tiles)
    const long wrow1 = wrow0 + (long)16 * K;

    f32x8 c00 = {}, c01 = {}, c10 = {}, c11 = {};

    for (int k0 = 0; k0 < K; k0 += 32) {
        // A 16x32 bf16 fragments: lane holds 16 contiguous K at k0 + 16h
        BfPack aH0, aH1, aL0, aL1, bH0, bH1, bL0, bL1;
        const long aoff0 = arow0 + k0 + 16 * half;
        const long aoff1 = arow1 + k0 + 16 * half;
        aH0.u[0] = *(const uint4*)(Ahi + aoff0);
        aH0.u[1] = *(const uint4*)(Ahi + aoff0 + 8);
        aL0.u[0] = *(const uint4*)(Alo + aoff0);
        aL0.u[1] = *(const uint4*)(Alo + aoff0 + 8);
        aH1.u[0] = *(const uint4*)(Ahi + aoff1);
        aH1.u[1] = *(const uint4*)(Ahi + aoff1 + 8);
        aL1.u[0] = *(const uint4*)(Alo + aoff1);
        aL1.u[1] = *(const uint4*)(Alo + aoff1 + 8);
        // B 32x16 bf16 fragments: lane holds 16 contiguous K at k0 + 16h
        const long woff0 = wrow0 + k0 + 16 * half;
        const long woff1 = wrow1 + k0 + 16 * half;
        bH0.u[0] = *(const uint4*)(Whi + woff0);
        bH0.u[1] = *(const uint4*)(Whi + woff0 + 8);
        bL0.u[0] = *(const uint4*)(Wlo + woff0);
        bL0.u[1] = *(const uint4*)(Wlo + woff0 + 8);
        bH1.u[0] = *(const uint4*)(Whi + woff1);
        bH1.u[1] = *(const uint4*)(Whi + woff1 + 8);
        bL1.u[0] = *(const uint4*)(Wlo + woff1);
        bL1.u[1] = *(const uint4*)(Wlo + woff1 + 8);

        if (k0 + 32 < K) {                    // L2-resident; hint next k-step
            __builtin_prefetch((const void*)(Ahi + aoff0 + 32), 0, 1);
            __builtin_prefetch((const void*)(Ahi + aoff1 + 32), 0, 1);
            __builtin_prefetch((const void*)(Whi + woff0 + 32), 0, 1);
            __builtin_prefetch((const void*)(Whi + woff1 + 32), 0, 1);
        }

        // bf16x3 split precision: hi*hi + hi*lo + lo*hi per tile (fp32 acc)
        c00 = __builtin_amdgcn_wmma_f32_16x16x32_bf16(false, aH0.v, false, bH0.v, (short)0, c00, false, false);
        c00 = __builtin_amdgcn_wmma_f32_16x16x32_bf16(false, aH0.v, false, bL0.v, (short)0, c00, false, false);
        c00 = __builtin_amdgcn_wmma_f32_16x16x32_bf16(false, aL0.v, false, bH0.v, (short)0, c00, false, false);

        c01 = __builtin_amdgcn_wmma_f32_16x16x32_bf16(false, aH0.v, false, bH1.v, (short)0, c01, false, false);
        c01 = __builtin_amdgcn_wmma_f32_16x16x32_bf16(false, aH0.v, false, bL1.v, (short)0, c01, false, false);
        c01 = __builtin_amdgcn_wmma_f32_16x16x32_bf16(false, aL0.v, false, bH1.v, (short)0, c01, false, false);

        c10 = __builtin_amdgcn_wmma_f32_16x16x32_bf16(false, aH1.v, false, bH0.v, (short)0, c10, false, false);
        c10 = __builtin_amdgcn_wmma_f32_16x16x32_bf16(false, aH1.v, false, bL0.v, (short)0, c10, false, false);
        c10 = __builtin_amdgcn_wmma_f32_16x16x32_bf16(false, aL1.v, false, bH0.v, (short)0, c10, false, false);

        c11 = __builtin_amdgcn_wmma_f32_16x16x32_bf16(false, aH1.v, false, bH1.v, (short)0, c11, false, false);
        c11 = __builtin_amdgcn_wmma_f32_16x16x32_bf16(false, aH1.v, false, bL1.v, (short)0, c11, false, false);
        c11 = __builtin_amdgcn_wmma_f32_16x16x32_bf16(false, aL1.v, false, bH1.v, (short)0, c11, false, false);
    }

    // D layout per tile: VGPR r -> row rowBase + r + 8*half, col = colBase + l15
    auto store_tile = [&](const f32x8& c, int rowBase, int col) {
        const float bv = bias ? bias[col] : 0.0f;
        #pragma unroll
        for (int r = 0; r < 8; ++r) {
            const int row = rowBase + r + 8 * half;
            const long idx = (long)row * N + col;
            float v = c[r] + bv;
            if (residual) v += residual[idx];
            C[idx] = v;
        }
    };
    store_tile(c00, bm * 32,      bn * 32 + l15);
    store_tile(c01, bm * 32,      bn * 32 + 16 + l15);
    store_tile(c10, bm * 32 + 16, bn * 32 + l15);
    store_tile(c11, bm * 32 + 16, bn * 32 + 16 + l15);
}

// ---------------------------------------------------------------------------
// Mask-token splice + ids_restore gather + positional embedding
// out x[b, s, d] for s in [0, 257)
// ---------------------------------------------------------------------------
__global__ void kd_splice_kernel(const float* __restrict__ proj,   // B x 65 x D
                                 const int*   __restrict__ ids,    // B x 256
                                 const float* __restrict__ mask_token, // D
                                 const float* __restrict__ pos,    // >=257 x D
                                 float* __restrict__ x,            // B x 257 x D
                                 int B, int L1, int D, int NP) {
    long i = (long)blockIdx.x * blockDim.x + threadIdx.x;
    long total = (long)B * (NP + 1) * D;
    if (i >= total) return;
    int d = (int)(i % D);
    long t = i / D;
    int s = (int)(t % (NP + 1));
    int b = (int)(t / (NP + 1));
    float v;
    if (s == 0) {
        v = proj[((long)b * L1) * D + d];
    } else {
        int id = ids[(long)b * NP + (s - 1)];
        v = (id < L1 - 1) ? proj[((long)b * L1 + 1 + id) * D + d] : mask_token[d];
    }
    x[i] = v + pos[(long)s * D + d];
}

// ---------------------------------------------------------------------------
// LayerNorm over last dim D, one row per block (256 threads)
// ---------------------------------------------------------------------------
__global__ void kd_layernorm_kernel(const float* __restrict__ x,
                                    const float* __restrict__ g,
                                    const float* __restrict__ b,
                                    float* __restrict__ y, int D) {
    const long row = blockIdx.x;
    const float* xr = x + row * D;
    float s = 0.f, ss = 0.f;
    for (int i = threadIdx.x; i < D; i += blockDim.x) {
        float v = xr[i]; s += v; ss += v * v;
    }
    #pragma unroll
    for (int o = 16; o > 0; o >>= 1) {
        s  += __shfl_down(s, o, 32);
        ss += __shfl_down(ss, o, 32);
    }
    __shared__ float rs[8], rss[8];
    __shared__ float stat[2];
    const int w = threadIdx.x >> 5;
    if ((threadIdx.x & 31) == 0) { rs[w] = s; rss[w] = ss; }
    __syncthreads();
    if (threadIdx.x == 0) {
        float S = 0.f, SS = 0.f;
        const int nw = blockDim.x >> 5;
        for (int i = 0; i < nw; ++i) { S += rs[i]; SS += rss[i]; }
        float m = S / D;
        float var = SS / D - m * m;
        stat[0] = m;
        stat[1] = rsqrtf(var + 1e-5f);
    }
    __syncthreads();
    const float m = stat[0], r = stat[1];
    for (int i = threadIdx.x; i < D; i += blockDim.x)
        y[row * D + i] = (xr[i] - m) * r * g[i] + b[i];
}

// ---------------------------------------------------------------------------
// MHA core: online-softmax attention per (batch, head). hd = 64.
// qkv: [B, S, 3D]; o: [B, S, D]
// ---------------------------------------------------------------------------
__global__ void kd_attention_kernel(const float* __restrict__ qkv,
                                    float* __restrict__ o,
                                    int B, int S, int D, int H) {
    const int hd = D / H;                     // 64
    const int bh = blockIdx.x;
    const int b = bh / H, h = bh % H;
    const long base = (long)b * S * (3 * D);
    const float scale = rsqrtf((float)hd);

    for (int q = threadIdx.x; q < S; q += blockDim.x) {
        float qreg[64];
        const float* qp = qkv + base + (long)q * (3 * D) + h * hd;
        #pragma unroll
        for (int d = 0; d < 64; ++d) qreg[d] = qp[d] * scale;

        float mx = -3.4e38f, l = 0.f;
        float acc[64];
        #pragma unroll
        for (int d = 0; d < 64; ++d) acc[d] = 0.f;

        for (int kk = 0; kk < S; ++kk) {
            const float* kp = qkv + base + (long)kk * (3 * D) + D + h * hd;
            float sv = 0.f;
            #pragma unroll
            for (int d = 0; d < 64; ++d) sv += qreg[d] * kp[d];
            const float nm = fmaxf(mx, sv);
            const float corr = __expf(mx - nm);
            const float p = __expf(sv - nm);
            l = l * corr + p;
            const float* vp = qkv + base + (long)kk * (3 * D) + 2 * D + h * hd;
            #pragma unroll
            for (int d = 0; d < 64; ++d) acc[d] = acc[d] * corr + p * vp[d];
            mx = nm;
        }
        const float inv = 1.f / l;
        float* op = o + ((long)b * S + q) * D + h * hd;
        #pragma unroll
        for (int d = 0; d < 64; ++d) op[d] = acc[d] * inv;
    }
}

// ---------------------------------------------------------------------------
// Cubic B-spline bases (uniform grid: G=5, k=3, [-1,1]) -> 8 values
// ---------------------------------------------------------------------------
__device__ __forceinline__ void kd_bspline8(float x, float* out8) {
    const float h = 0.4f, lo = -1.0f;
    float b[11];
    #pragma unroll
    for (int i = 0; i < 11; ++i) {
        float g0 = lo + (i - 3) * h;
        b[i] = (x >= g0 && x < g0 + h) ? 1.f : 0.f;
    }
    #pragma unroll
    for (int j = 1; j <= 3; ++j) {
        const float inv = 1.0f / (j * h);
        #pragma unroll
        for (int i = 0; i < 10; ++i) {
            if (i < 11 - j) {
                float gi = lo + (i - 3) * h;
                float gr = lo + (i + j + 1 - 3) * h;
                b[i] = (x - gi) * inv * b[i] + (gr - x) * inv * b[i + 1];
            }
        }
    }
    #pragma unroll
    for (int i = 0; i < 8; ++i) out8[i] = b[i];
}

__device__ __forceinline__ float kd_silu(float v) {
    return v / (1.f + __expf(-v));
}

// ---------------------------------------------------------------------------
// KAN layer 1: [M,512] -> [M,8]. One wave per row, lane-strided over inputs,
// cross-lane reduction via shfl (wave32).
// ---------------------------------------------------------------------------
__global__ void kd_kan1_kernel(const float* __restrict__ x,      // M x IN
                               const float* __restrict__ bw,     // 8 x IN
                               const float* __restrict__ spw,    // 8 x IN x 8
                               const float* __restrict__ scl,    // 8 x IN
                               float* __restrict__ y,            // M x 8
                               int M, int IN) {
    const int wave = (int)((blockIdx.x * blockDim.x + threadIdx.x) >> 5);
    const int lane = threadIdx.x & 31;
    if (wave >= M) return;
    const float* xr = x + (long)wave * IN;
    float acc[8] = {0.f, 0.f, 0.f, 0.f, 0.f, 0.f, 0.f, 0.f};
    for (int i = lane; i < IN; i += 32) {
        const float v = xr[i];
        const float sil = kd_silu(v);
        float bs[8];
        kd_bspline8(v, bs);
        #pragma unroll
        for (int o = 0; o < 8; ++o) {
            const long wi = (long)o * IN + i;
            const float* sp = spw + wi * 8;
            float t = 0.f;
            #pragma unroll
            for (int j = 0; j < 8; ++j) t += bs[j] * sp[j];
            acc[o] += sil * bw[wi] + t * scl[wi];
        }
    }
    #pragma unroll
    for (int o = 0; o < 8; ++o)
        #pragma unroll
        for (int off = 16; off > 0; off >>= 1)
            acc[o] += __shfl_down(acc[o], off, 32);
    if (lane == 0) {
        #pragma unroll
        for (int o = 0; o < 8; ++o) y[(long)wave * 8 + o] = acc[o];
    }
}

// ---------------------------------------------------------------------------
// KAN layer 2: [M,8] -> out [B,256,256] (token 0 dropped). One row per block;
// 72-entry feature vector in LDS, one output per thread.
// ---------------------------------------------------------------------------
__global__ void kd_kan2_kernel(const float* __restrict__ y1,   // M x 8
                               const float* __restrict__ bw,   // 256 x 8
                               const float* __restrict__ spw,  // 256 x 8 x 8
                               const float* __restrict__ scl,  // 256 x 8
                               float* __restrict__ out,        // B x 256 x 256
                               int S, int OUT) {
    const int row = blockIdx.x;          // 0 .. M-1
    const int s = row % S, b = row / S;
    __shared__ float feat[8 + 64];
    if (threadIdx.x < 8) {
        float v = y1[(long)row * 8 + threadIdx.x];
        feat[threadIdx.x] = kd_silu(v);
        float bs[8];
        kd_bspline8(v, bs);
        #pragma unroll
        for (int j = 0; j < 8; ++j) feat[8 + threadIdx.x * 8 + j] = bs[j];
    }
    __syncthreads();
    if (s == 0) return;                  // token 0 is sliced away
    const int o = threadIdx.x;
    float acc = 0.f;
    #pragma unroll
    for (int i = 0; i < 8; ++i) {
        const int wi = o * 8 + i;
        acc += feat[i] * bw[wi];
        const float sc = scl[wi];
        const float* sp = spw + (long)wi * 8;
        #pragma unroll
        for (int j = 0; j < 8; ++j) acc += feat[8 + i * 8 + j] * sp[j] * sc;
    }
    out[(((long)b * (S - 1)) + (s - 1)) * OUT + o] = acc;
}

// ---------------------------------------------------------------------------
// Host orchestration
// ---------------------------------------------------------------------------
extern "C" void kernel_launch(void* const* d_in, const int* in_sizes, int n_in,
                              void* d_out, int out_size, void* d_ws, size_t ws_size,
                              hipStream_t stream) {
    const float* x_in       = (const float*)d_in[0];
    const int*   ids        = (const int*)  d_in[1];
    const float* dec_w      = (const float*)d_in[2];
    const float* dec_b      = (const float*)d_in[3];
    const float* mask_token = (const float*)d_in[4];
    const float* pos        = (const float*)d_in[5];
    const float* ln1_g      = (const float*)d_in[6];
    const float* ln1_b      = (const float*)d_in[7];
    const float* in_w       = (const float*)d_in[8];
    const float* in_b       = (const float*)d_in[9];
    const float* out_w      = (const float*)d_in[10];
    const float* out_b      = (const float*)d_in[11];
    const float* ln2_g      = (const float*)d_in[12];
    const float* ln2_b      = (const float*)d_in[13];
    const float* k1_base    = (const float*)d_in[14];
    const float* k1_spline  = (const float*)d_in[15];
    const float* k1_scaler  = (const float*)d_in[16];
    const float* k2_base    = (const float*)d_in[17];
    const float* k2_spline  = (const float*)d_in[18];
    const float* k2_scaler  = (const float*)d_in[19];

    constexpr int B = 64, L1 = 65, E = 384, D = 512, S = 257, H = 8;
    constexpr int OUT = 256, NP = 256;
    constexpr int M0 = B * L1;   // 4160 rows into projection
    constexpr int M  = B * S;    // 16448 token rows

    // ---- workspace carve-out (256B aligned) ----
    char* ws = (char*)d_ws;
    size_t off = 0;
    auto take = [&](size_t bytes) -> char* {
        char* p = ws + off;
        off = (off + bytes + 255) & ~(size_t)255;
        return p;
    };
    unsigned short* a_hi = (unsigned short*)take((size_t)M * D * 2);      // act splits
    unsigned short* a_lo = (unsigned short*)take((size_t)M * D * 2);
    unsigned short* w_hi = (unsigned short*)take((size_t)3 * D * D * 2);  // weight splits
    unsigned short* w_lo = (unsigned short*)take((size_t)3 * D * D * 2);
    float* proj = (float*)take((size_t)M0 * D * 4);
    float* xres = (float*)take((size_t)M * D * 4);
    float* tmp  = (float*)take((size_t)M * D * 4);
    float* qkv  = (float*)take((size_t)M * 3 * D * 4);
    float* xres2 = qkv;                       // qkv is dead after attention
    float* y1    = qkv + (size_t)M * D;       // lives in qkv tail

    auto split = [&](const float* src, unsigned short* h, unsigned short* l, long n) {
        kd_split_kernel<<<dim3((unsigned)((n + 255) / 256)), dim3(256), 0, stream>>>(src, h, l, n);
    };
    auto gemm = [&](const unsigned short* Ah, const unsigned short* Al,
                    const unsigned short* Wh, const unsigned short* Wl,
                    const float* bias, const float* resid, float* Cout,
                    int Mm, int Nn, int Kk) {
        const int blocks32 = (Mm / 32) * (Nn / 32);   // one 32x32 block per wave
        kd_gemm_bf16x3<<<dim3((unsigned)((blocks32 + 3) / 4)), dim3(128), 0, stream>>>(
            Ah, Al, Wh, Wl, bias, resid, Cout, Mm, Nn, Kk);
    };

    // 1) projection: proj = x @ dec_w^T + dec_b          [4160 x 384 x 512]
    split(x_in, a_hi, a_lo, (long)M0 * E);
    split(dec_w, w_hi, w_lo, (long)D * E);
    gemm(a_hi, a_lo, w_hi, w_lo, dec_b, nullptr, proj, M0, D, E);

    // 2) mask-token splice + restore + pos embed -> xres [M x D]
    {
        long n = (long)M * D;
        kd_splice_kernel<<<dim3((unsigned)((n + 255) / 256)), dim3(256), 0, stream>>>(
            proj, ids, mask_token, pos, xres, B, L1, D, NP);
    }

    // 3) LN1 -> QKV GEMM -> attention -> out-proj (+residual)
    kd_layernorm_kernel<<<dim3(M), dim3(256), 0, stream>>>(xres, ln1_g, ln1_b, tmp, D);
    split(tmp, a_hi, a_lo, (long)M * D);
    split(in_w, w_hi, w_lo, (long)3 * D * D);
    gemm(a_hi, a_lo, w_hi, w_lo, in_b, nullptr, qkv, M, 3 * D, D);

    kd_attention_kernel<<<dim3(B * H), dim3(256), 0, stream>>>(qkv, tmp, B, S, D, H);

    split(tmp, a_hi, a_lo, (long)M * D);
    split(out_w, w_hi, w_lo, (long)D * D);
    gemm(a_hi, a_lo, w_hi, w_lo, out_b, xres, xres2, M, D, D);

    // 4) LN2 -> KAN(512->8) -> KAN(8->256) -> d_out
    kd_layernorm_kernel<<<dim3(M), dim3(256), 0, stream>>>(xres2, ln2_g, ln2_b, tmp, D);
    kd_kan1_kernel<<<dim3((M + 7) / 8), dim3(256), 0, stream>>>(
        tmp, k1_base, k1_spline, k1_scaler, y1, M, D);
    kd_kan2_kernel<<<dim3(M), dim3(256), 0, stream>>>(
        y1, k2_base, k2_spline, k2_scaler, (float*)d_out, S, OUT);
}
